// TemplatePairEmbedderMultimer_17832704213164
// MI455X (gfx1250) — matrix operand order repaired and checked by reading, not compiled, not run on hardware
//
#include <hip/hip_runtime.h>

typedef __attribute__((ext_vector_type(16))) _Float16 v16h;
typedef __attribute__((ext_vector_type(8)))  float    v8f;
typedef __attribute__((ext_vector_type(4)))  float    v4f;

#define NRES  768
#define CIN   128
#define COUT  64
#define CDG   39
#define CAA   22
#define CESM  1280

// workspace layout (bytes)
#define COL_OFF   0                              // 768*64 f32 col_add (aa1+esm1+biases)
#define ROW_OFF   (COL_OFF + NRES*COUT*4)        // 768*64 f32 row_add (aa2+esm2+biases)
#define BIAS_OFF  (ROW_OFF + NRES*COUT*4)        // 64 f32 folded biases (padded to 256B)
#define BQ_OFF    (BIAS_OFF + 256)               // 4 ntiles * 4 ksteps * 32 lanes * 16 f16
#define BD_OFF    (BQ_OFF + 4*4*32*16*2)         // 4 ntiles * 2 ksteps * 32 lanes * 16 f16
// total = 418048 bytes

// ---------------------------------------------------------------------------
// Pack w_q / w_dgram into CDNA5 WMMA B-operand lane layout; fold biases.
// B layout (16-bit, 32x16): lane n = lane&15, hi = lane>>4,
// halves 0..7 -> K=kstep*32+hi*8+h, halves 8..15 -> K=kstep*32+16+hi*8+(h-8).
// ---------------------------------------------------------------------------
__global__ void pack_b_kernel(const float* __restrict__ wq, const float* __restrict__ wd,
                              const float* __restrict__ b_dgram, const float* __restrict__ b_pbm,
                              const float* __restrict__ b_x, const float* __restrict__ b_y,
                              const float* __restrict__ b_z, const float* __restrict__ b_bb,
                              const float* __restrict__ b_q, char* __restrict__ ws) {
  _Float16* Bq  = (_Float16*)(ws + BQ_OFF);
  _Float16* Bd  = (_Float16*)(ws + BD_OFF);
  float* bias   = (float*)(ws + BIAS_OFF);
  int tid = threadIdx.x;                         // 512 threads
  {
    int t = tid >> 7, s = (tid >> 5) & 3, lane = tid & 31;
    int n = t * 16 + (lane & 15);
    int khi = (lane < 16) ? 0 : 16;
    for (int h = 0; h < 16; ++h) {
      int K = s * 32 + (khi >> 1) + (h & 7) + ((h >> 3) << 4);
      Bq[((t * 4 + s) * 32 + lane) * 16 + h] = (_Float16)wq[K * COUT + n];
    }
  }
  if (tid < 256) {
    int t = tid >> 6, s = (tid >> 5) & 1, lane = tid & 31;
    int n = t * 16 + (lane & 15);
    int khi = (lane < 16) ? 0 : 16;
    for (int h = 0; h < 16; ++h) {
      int K = s * 32 + (khi >> 1) + (h & 7) + ((h >> 3) << 4);
      float v = (K < CDG) ? wd[K * COUT + n] : 0.0f;
      Bd[((t * 2 + s) * 32 + lane) * 16 + h] = (_Float16)v;
    }
  }
  if (tid < COUT) {
    bias[tid] = b_dgram[tid] + b_pbm[tid] + b_x[tid] + b_y[tid] +
                b_z[tid] + b_bb[tid] + b_q[tid];
  }
}

// ---------------------------------------------------------------------------
// Per-residue projections broadcast over rows / columns:
//   col_add[j][o] = aa[j]@w_aa1 + b_aa1 + esm[j]@w_esm1 + b_esm1
//   row_add[i][o] = aa[i]@w_aa2 + b_aa2 + esm[i]@w_esm2 + b_esm2
// ---------------------------------------------------------------------------
__global__ void rowcol_kernel(const float* __restrict__ aa, const float* __restrict__ esm,
                              const float* __restrict__ w_aa1, const float* __restrict__ b_aa1,
                              const float* __restrict__ w_aa2, const float* __restrict__ b_aa2,
                              const float* __restrict__ w_e1, const float* __restrict__ b_e1,
                              const float* __restrict__ w_e2, const float* __restrict__ b_e2,
                              char* __restrict__ ws) {
  __shared__ float es[CESM];
  __shared__ float as[CAA];
  int r = blockIdx.x, o = threadIdx.x;           // 64 threads
  for (int k = o; k < CESM; k += 64) es[k] = esm[r * CESM + k];
  if (o < CAA) as[o] = aa[r * CAA + o];
  __syncthreads();
  float a1 = b_aa1[o] + b_e1[o];
  float a2 = b_aa2[o] + b_e2[o];
  for (int k = 0; k < CAA; ++k)  { a1 += as[k] * w_aa1[k * COUT + o]; a2 += as[k] * w_aa2[k * COUT + o]; }
  for (int k = 0; k < CESM; ++k) { a1 += es[k] * w_e1[k * COUT + o];  a2 += es[k] * w_e2[k * COUT + o]; }
  ((float*)(ws + COL_OFF))[r * COUT + o] = a1;
  ((float*)(ws + ROW_OFF))[r * COUT + o] = a2;
}

// ---------------------------------------------------------------------------
// Main fused kernel: one wave per 16-pixel tile (row i, cols jb..jb+15).
// A-operand lane layout (16-bit, 16x32): lane m=lane&15, hi=lane>>4,
// halves 0..7 -> K = s*32 + hi*8 + h, halves 8..15 -> K = s*32 + 16 + hi*8 + (h-8).
// Each lane therefore owns K-blocks (2c+hi)*8..+7 for c=0..7 (query) / 0..3 (dgram).
// ---------------------------------------------------------------------------
__global__ void __launch_bounds__(128) fused_pair_kernel(
    const float* __restrict__ td, const float* __restrict__ q,
    const float* __restrict__ pbm, const float* __restrict__ bbm,
    const float* __restrict__ mc2, const float* __restrict__ ux,
    const float* __restrict__ uy, const float* __restrict__ uz,
    const float* __restrict__ lng, const float* __restrict__ lnb,
    const float* __restrict__ w_pbm, const float* __restrict__ w_x,
    const float* __restrict__ w_y, const float* __restrict__ w_z,
    const float* __restrict__ w_bb,
    const char* __restrict__ ws, float* __restrict__ out) {
  const int lane = threadIdx.x & 31;
  const int wv   = threadIdx.x >> 5;
  const int gw   = blockIdx.x * 4 + wv;          // tile id, 0..36863
  const int i    = gw / (NRES / 16);
  const int jb   = (gw % (NRES / 16)) * 16;
  const int hi   = lane >> 4;
  const int m    = lane & 15;
  const long p   = (long)i * NRES + jb + m;      // this lane's owned pixel

  // per-pixel scalar branches
  const float mcv = mc2[p];
  const float bb2 = bbm[i] * bbm[jb + m] * mcv;
  const float pb2 = pbm[i] * pbm[jb + m] * mcv;
  const float xs = ux[p] * bb2, ys = uy[p] * bb2, zs = uz[p] * bb2;

  // ---- dgram A operand (K padded 39 -> 64), masked+scaled at load ----
  v16h aD[2];
  const float* tdp = td + p * CDG;
  #pragma unroll
  for (int c = 0; c < 4; ++c) {
    #pragma unroll
    for (int e = 0; e < 8; ++e) {
      int K = (2 * c + hi) * 8 + e;
      float v = 0.0f;
      if (K < CDG) v = tdp[K] * bb2;
      aD[c >> 1][(c & 1) * 8 + e] = (_Float16)v;
    }
  }

  // ---- query LayerNorm (each lane owns 64 of 128 channels; partner = lane^16) ----
  const float* qp = q + p * CIN;
  float qv[64];
  float sum = 0.0f, sq = 0.0f;
  #pragma unroll
  for (int c = 0; c < 8; ++c) {
    const v4f* src4 = (const v4f*)(qp + (2 * c + hi) * 8);
    v4f v0 = src4[0], v1 = src4[1];
    #pragma unroll
    for (int e = 0; e < 4; ++e) {
      qv[c * 8 + e]     = v0[e];
      qv[c * 8 + 4 + e] = v1[e];
      sum += v0[e] + v1[e];
      sq  += v0[e] * v0[e] + v1[e] * v1[e];
    }
  }
  sum += __shfl_xor(sum, 16, 32);
  sq  += __shfl_xor(sq, 16, 32);
  const float mean = sum * (1.0f / (float)CIN);
  const float var  = sq * (1.0f / (float)CIN) - mean * mean;
  const float inv  = rsqrtf(var + 1e-5f);

  v16h aQ[4];
  #pragma unroll
  for (int c = 0; c < 8; ++c) {
    #pragma unroll
    for (int e = 0; e < 8; ++e) {
      int K = (2 * c + hi) * 8 + e;
      float vn = (qv[c * 8 + e] - mean) * inv * lng[K] + lnb[K];
      aQ[c >> 1][(c & 1) * 8 + e] = (_Float16)vn;
    }
  }

  // ---- WMMA: both GEMMs accumulate into the same f32 C ----
  const v16h* Bq = (const v16h*)(ws + BQ_OFF);
  const v16h* Bd = (const v16h*)(ws + BD_OFF);
  const float* bias = (const float*)(ws + BIAS_OFF);
  const float* col  = (const float*)(ws + COL_OFF);
  const float* row  = (const float*)(ws + ROW_OFF);

  v8f acc[4];
  #pragma unroll
  for (int t = 0; t < 4; ++t) {
    v8f c8 = {0.f, 0.f, 0.f, 0.f, 0.f, 0.f, 0.f, 0.f};
    #pragma unroll
    for (int s = 0; s < 2; ++s)
      c8 = __builtin_amdgcn_wmma_f32_16x16x32_f16(
          false, aD[s], false, Bd[(t * 2 + s) * 32 + lane], (short)0, c8, false, false);
    #pragma unroll
    for (int s = 0; s < 4; ++s)
      c8 = __builtin_amdgcn_wmma_f32_16x16x32_f16(
          false, aQ[s], false, Bq[(t * 4 + s) * 32 + lane], (short)0, c8, false, false);
    acc[t] = c8;
  }

  // ---- epilogue: C/D layout is (VGPR r, lane) -> M = r + 8*hi, N = t*16 + m ----
  #pragma unroll
  for (int t = 0; t < 4; ++t) {
    const int o = t * 16 + m;
    const float wp = w_pbm[o], wxv = w_x[o], wyv = w_y[o], wzv = w_z[o], wbv = w_bb[o];
    const float bt = bias[o] + row[i * COUT + o];
    #pragma unroll
    for (int r = 0; r < 8; ++r) {
      const int srcl = r + hi * 8;               // pixel index within tile
      float val = acc[t][r]
                + __shfl(pb2, srcl, 32) * wp
                + __shfl(xs,  srcl, 32) * wxv
                + __shfl(ys,  srcl, 32) * wyv
                + __shfl(zs,  srcl, 32) * wzv
                + __shfl(bb2, srcl, 32) * wbv
                + bt + col[(jb + srcl) * COUT + o];
      out[((long)i * NRES + jb + srcl) * (long)COUT + o] = val;
    }
  }
}

// ---------------------------------------------------------------------------
extern "C" void kernel_launch(void* const* d_in, const int* in_sizes, int n_in,
                              void* d_out, int out_size, void* d_ws, size_t ws_size,
                              hipStream_t stream) {
  const float* td   = (const float*)d_in[0];
  const float* aa   = (const float*)d_in[1];
  const float* q    = (const float*)d_in[2];
  const float* pbm  = (const float*)d_in[3];
  const float* bbm  = (const float*)d_in[4];
  const float* mc2  = (const float*)d_in[5];
  const float* ux   = (const float*)d_in[6];
  const float* uy   = (const float*)d_in[7];
  const float* uz   = (const float*)d_in[8];
  const float* esm  = (const float*)d_in[9];
  const float* w_dg = (const float*)d_in[10];
  const float* b_dg = (const float*)d_in[11];
  const float* w_a1 = (const float*)d_in[12];
  const float* b_a1 = (const float*)d_in[13];
  const float* w_a2 = (const float*)d_in[14];
  const float* b_a2 = (const float*)d_in[15];
  const float* lng  = (const float*)d_in[16];
  const float* lnb  = (const float*)d_in[17];
  const float* w_q  = (const float*)d_in[18];
  const float* b_q  = (const float*)d_in[19];
  const float* w_pb = (const float*)d_in[20];
  const float* b_pb = (const float*)d_in[21];
  const float* w_x  = (const float*)d_in[22];
  const float* b_x  = (const float*)d_in[23];
  const float* w_y  = (const float*)d_in[24];
  const float* b_y  = (const float*)d_in[25];
  const float* w_z  = (const float*)d_in[26];
  const float* b_z  = (const float*)d_in[27];
  const float* w_bb = (const float*)d_in[28];
  const float* b_bb = (const float*)d_in[29];
  const float* w_e1 = (const float*)d_in[30];
  const float* b_e1 = (const float*)d_in[31];
  const float* w_e2 = (const float*)d_in[32];
  const float* b_e2 = (const float*)d_in[33];
  char* ws = (char*)d_ws;

  pack_b_kernel<<<1, 512, 0, stream>>>(w_q, w_dg, b_dg, b_pb, b_x, b_y, b_z, b_bb, b_q, ws);
  rowcol_kernel<<<NRES, 64, 0, stream>>>(aa, esm, w_a1, b_a1, w_a2, b_a2, w_e1, b_e1, w_e2, b_e2, ws);

  const int tiles = (NRES * NRES) / 16;          // 36864 waves
  fused_pair_kernel<<<tiles / 4, 128, 0, stream>>>(
      td, q, pbm, bbm, mc2, ux, uy, uz, lng, lnb,
      w_pb, w_x, w_y, w_z, w_bb, ws, (float*)d_out);
}
